// MultiheadAttention_1511828488777
// MI455X (gfx1250) — matrix operand order repaired and checked by reading, not compiled, not run on hardware
//
#include <hip/hip_runtime.h>
#include <hip/hip_bf16.h>
#include <stdint.h>

// ---------------------------------------------------------------------------
// MHA forward for MI455X (gfx1250, wave32, WMMA).
// bf16 WMMA (v_wmma_f32_16x16x32_bf16) with f32 accumulation for all GEMMs;
// two-pass flash-style attention so the 537MB (B,H,T,S) score tensor never
// touches HBM; head-averaged attention accumulated with global f32 atomics
// into an L2-resident 33MB buffer.  K/V tile staging uses gfx1250
// GLOBAL_LOAD_ASYNC_TO_LDS_B128 (ASYNCcnt) when the builtin is available.
// ---------------------------------------------------------------------------

typedef __bf16 bf16;
typedef __attribute__((ext_vector_type(16))) __bf16 v16bf;
typedef __attribute__((ext_vector_type(8)))  float  v8f;

constexpr int T_ = 2048, B_ = 2, E_ = 1024, H_ = 16, HD_ = 64;
constexpr int M_ = T_ * B_;                  // 4096 rows in projection GEMMs
constexpr int BM = 128, BN = 128, BK = 32;   // GEMM block tile

__device__ __forceinline__ v8f wmma_bf16(v16bf a, v16bf b, v8f c) {
  return __builtin_amdgcn_wmma_f32_16x16x32_bf16(false, a, false, b,
                                                 (short)0, c, false, false);
}

// ---- gfx1250 async global->LDS staging (guarded; sync fallback) -----------
#if defined(__AMDGCN__) && __has_builtin(__builtin_amdgcn_global_load_async_to_lds_b128)
#define ATTN_ASYNC_LDS 1
typedef int v4i __attribute__((__vector_size__(4 * sizeof(int))));
typedef __attribute__((address_space(1))) v4i gv4i;  // global (printed __device__)
typedef __attribute__((address_space(3))) v4i sv4i;  // LDS
__device__ __forceinline__ void async_cp16(const void* g, void* l) {
  // generic->AS casts via uintptr_t: global bits identical; LDS offset is the
  // low 32 bits of the generic address (aperture rule), truncated by AS3 ptr.
  __builtin_amdgcn_global_load_async_to_lds_b128(
      (gv4i*)(uintptr_t)g, (sv4i*)(uintptr_t)l, 0, 0);
}
__device__ __forceinline__ void wait_async0() {
#if __has_builtin(__builtin_amdgcn_s_wait_asynccnt)
  __builtin_amdgcn_s_wait_asynccnt(0);
#else
  asm volatile("s_wait_asynccnt 0x0" ::: "memory");
#endif
}
#endif

// ---- WMMA fragment loaders, wave32 layouts per CDNA5 ISA §7.12.2 ----------
// A (16x32, M x K): lanes 0-15 = rows 0-15, lanes 16-31 same rows with K+8;
// VGPR v holds K pair {k0,k0+1}, k0 = (v<4 ? 2v : 16+2(v-4)) + 8*(lane>=16).
__device__ __forceinline__ v16bf load_a_frag_lds(const bf16* tile, int ldk,
                                                 int l16, int lh) {
  v16bf a;
#pragma unroll
  for (int v = 0; v < 8; ++v) {
    const int kk = ((v < 4) ? 2 * v : 16 + 2 * (v - 4)) + lh * 8;
    const bf16* p = tile + l16 * ldk + kk;
    a[2 * v]     = p[0];
    a[2 * v + 1] = p[1];
  }
  return a;
}
// B (32x16, K x N), tile stored [N][K] (row = n, contiguous K):
// lanes 0-15 = cols 0-15 with K 0-7,16-23; lanes 16-31 same cols with K 8-15,24-31.
__device__ __forceinline__ v16bf load_b_frag_lds(const bf16* tile, int ldk,
                                                 int l16, int lh) {
  v16bf b;
#pragma unroll
  for (int v = 0; v < 8; ++v) {
    const int kk = ((v < 4) ? 2 * v : 16 + 2 * (v - 4)) + lh * 8;
    const bf16* p = tile + l16 * ldk + kk;
    b[2 * v]     = p[0];
    b[2 * v + 1] = p[1];
  }
  return b;
}

// ---------------------------------------------------------------------------
// C(M,N) = (A(M,K) @ W(N,K)^T + bias) * scale.  A is f32 or bf16 (converted
// to bf16 while staging into LDS).  8 waves as 4(M) x 2(N); each wave owns a
// 32x64 tile: 2 A-frags + 4 B-frags feed 8 WMMAs per K-step.
// ---------------------------------------------------------------------------
template <typename AT, typename OT>
__global__ __launch_bounds__(256) void gemm_xwT_bias(
    const AT* __restrict__ A, const float* __restrict__ W,
    const float* __restrict__ bias, OT* __restrict__ C, float scale) {
  constexpr int LDA = BK + 2;  // pad: 17-dword row stride, conflict-free
  __shared__ bf16 As[BM][LDA];
  __shared__ bf16 Bs[BN][LDA];

  const int tid  = threadIdx.x;
  const int lane = tid & 31;
  const int wave = tid >> 5;
  const int wm   = wave & 3;   // 4 M tiles of 32
  const int wn   = wave >> 2;  // 2 N tiles of 64
  const int l16  = lane & 15;
  const int lh   = lane >> 4;
  const int m0   = blockIdx.y * BM;
  const int n0   = blockIdx.x * BN;

  v8f acc[2][4] = {};

  for (int k0 = 0; k0 < E_; k0 += BK) {
    // stage A tile (128x32) and W tile (128x32): 16 elems/thread each
    {
      const int r  = tid >> 1;
      const int cc = (tid & 1) * 16;
      const AT* srcA = A + (size_t)(m0 + r) * E_ + k0 + cc;
      const float* srcW = W + (size_t)(n0 + r) * E_ + k0 + cc;
#pragma unroll
      for (int j = 0; j < 16; ++j) As[r][cc + j] = (bf16)(float)srcA[j];
#pragma unroll
      for (int j = 0; j < 16; ++j) Bs[r][cc + j] = (bf16)srcW[j];
      if (k0 + BK < E_) {  // gfx1250 global_prefetch_b8 for next K tiles
        __builtin_prefetch(srcA + BK, 0, 1);
        __builtin_prefetch(srcW + BK, 0, 1);
      }
    }
    __syncthreads();

    const v16bf a0 = load_a_frag_lds(&As[wm * 32 + 0][0],  LDA, l16, lh);
    const v16bf a1 = load_a_frag_lds(&As[wm * 32 + 16][0], LDA, l16, lh);
    v16bf bfr[4];
#pragma unroll
    for (int j = 0; j < 4; ++j)
      bfr[j] = load_b_frag_lds(&Bs[wn * 64 + j * 16][0], LDA, l16, lh);
#pragma unroll
    for (int j = 0; j < 4; ++j) {
      acc[0][j] = wmma_bf16(a0, bfr[j], acc[0][j]);
      acc[1][j] = wmma_bf16(a1, bfr[j], acc[1][j]);
    }
    __syncthreads();
  }

  // epilogue: C-fragment layout (lane = N, VGPR v -> M = v + 8*(lane>=16))
#pragma unroll
  for (int i = 0; i < 2; ++i)
#pragma unroll
    for (int j = 0; j < 4; ++j) {
      const int n    = n0 + wn * 64 + j * 16 + l16;
      const float bb = bias[n];
#pragma unroll
      for (int v = 0; v < 8; ++v) {
        const int m = m0 + wm * 32 + i * 16 + v + lh * 8;
        C[(size_t)m * E_ + n] = (OT)((acc[i][j][v] + bb) * scale);
      }
    }
}

// ---------------------------------------------------------------------------
// Two-pass attention per (b, h, 64 query rows).  4 waves x 16 rows each.
// Pass 1: online row max / exp-sum over S.  Pass 2: normalized P, atomic
// head-average accumulation, and P@V via WMMA (P re-laid-out through LDS).
// ---------------------------------------------------------------------------
__global__ __launch_bounds__(128) void attn_fwd(
    const bf16* __restrict__ Q, const bf16* __restrict__ K,
    const bf16* __restrict__ V, bf16* __restrict__ O,
    float* __restrict__ avg) {
  constexpr int LDS_K = 72;  // bf16 row stride (144B: uint4/b128-aligned)
  __shared__ bf16 Ks[64][LDS_K];
  __shared__ bf16 Vs[64][LDS_K];
  __shared__ bf16 Ps[4][16][LDS_K];

  const int tid  = threadIdx.x;
  const int lane = tid & 31;
  const int wave = tid >> 5;
  const int l16  = lane & 15;
  const int lh   = lane >> 4;
  const int h    = blockIdx.y;
  const int b    = blockIdx.z;
  const int t0   = blockIdx.x * 64 + wave * 16;

  const size_t rowStride = (size_t)B_ * E_;                 // (T,B,E) row hop
  const size_t headOff   = (size_t)b * E_ + (size_t)h * HD_;

  // Q A-fragments for this wave's 16 rows: two K=32 chunks over d (HD=64)
  v16bf qf[2];
  {
    const bf16* qrow = Q + (size_t)(t0 + l16) * rowStride + headOff;
#pragma unroll
    for (int dc = 0; dc < 2; ++dc)
#pragma unroll
      for (int v = 0; v < 8; ++v) {
        const int kk = dc * 32 + ((v < 4) ? 2 * v : 16 + 2 * (v - 4)) + lh * 8;
        qf[dc][2 * v]     = qrow[kk];
        qf[dc][2 * v + 1] = qrow[kk + 1];
      }
  }

  float mrow[8], lrow[8];
#pragma unroll
  for (int v = 0; v < 8; ++v) { mrow[v] = -1e30f; lrow[v] = 0.0f; }

  const int r  = tid >> 1;       // staging: row within 64-row chunk
  const int hf = tid & 1;        // staging: which 32-element half row

  // ---------------- pass 1: row max / sum over all S ----------------
  for (int s0 = 0; s0 < T_; s0 += 64) {
    __syncthreads();
    {  // cooperative stage: K rows s0..s0+63, 64 bf16 per row
      const bf16* src = K + (size_t)(s0 + r) * rowStride + headOff + hf * 32;
      bf16* dst = &Ks[r][hf * 32];
#ifdef ATTN_ASYNC_LDS
#pragma unroll
      for (int j = 0; j < 4; ++j) async_cp16(src + j * 8, dst + j * 8);
      wait_async0();
#else
      const uint4* s4 = (const uint4*)src;
      uint4* d4 = (uint4*)dst;
#pragma unroll
      for (int j = 0; j < 4; ++j) d4[j] = s4[j];
#endif
    }
    __syncthreads();

    float sc[4][8];
    float cmax[8];
#pragma unroll
    for (int v = 0; v < 8; ++v) cmax[v] = -1e30f;
#pragma unroll
    for (int nt = 0; nt < 4; ++nt) {  // 16 S-columns per tile
      v8f c = {};
#pragma unroll
      for (int dc = 0; dc < 2; ++dc) {
        const v16bf kf = load_b_frag_lds(&Ks[nt * 16][dc * 32], LDS_K, l16, lh);
        c = wmma_bf16(qf[dc], kf, c);
      }
#pragma unroll
      for (int v = 0; v < 8; ++v) {
        sc[nt][v] = c[v];
        cmax[v]   = fmaxf(cmax[v], c[v]);
      }
    }
    // reduce across the 16-lane half-groups (N dimension of C fragment)
#pragma unroll
    for (int off = 1; off < 16; off <<= 1)
#pragma unroll
      for (int v = 0; v < 8; ++v)
        cmax[v] = fmaxf(cmax[v], __shfl_xor(cmax[v], off, 32));
    float se[8];
#pragma unroll
    for (int v = 0; v < 8; ++v) {
      const float nm = fmaxf(mrow[v], cmax[v]);
      float s = 0.0f;
#pragma unroll
      for (int nt = 0; nt < 4; ++nt) s += __expf(sc[nt][v] - nm);
      se[v]   = s;
      cmax[v] = nm;
    }
#pragma unroll
    for (int off = 1; off < 16; off <<= 1)
#pragma unroll
      for (int v = 0; v < 8; ++v) se[v] += __shfl_xor(se[v], off, 32);
#pragma unroll
    for (int v = 0; v < 8; ++v) {
      lrow[v] = lrow[v] * __expf(mrow[v] - cmax[v]) + se[v];
      mrow[v] = cmax[v];
    }
  }

  float invl[8];
#pragma unroll
  for (int v = 0; v < 8; ++v) invl[v] = 1.0f / lrow[v];

  // ---------------- pass 2: P, head-average, P@V ----------------
  v8f o[4] = {};
  const float invH = 1.0f / (float)H_;
  float* avgBase = avg + (size_t)b * T_ * T_;

  for (int s0 = 0; s0 < T_; s0 += 64) {
    __syncthreads();
    {  // stage K and V chunks
      const bf16* srcK = K + (size_t)(s0 + r) * rowStride + headOff + hf * 32;
      const bf16* srcV = V + (size_t)(s0 + r) * rowStride + headOff + hf * 32;
      bf16* dk = &Ks[r][hf * 32];
      bf16* dv = &Vs[r][hf * 32];
#ifdef ATTN_ASYNC_LDS
#pragma unroll
      for (int j = 0; j < 4; ++j) {
        async_cp16(srcK + j * 8, dk + j * 8);
        async_cp16(srcV + j * 8, dv + j * 8);
      }
      wait_async0();
#else
      const uint4* k4 = (const uint4*)srcK;
      const uint4* v4 = (const uint4*)srcV;
#pragma unroll
      for (int j = 0; j < 4; ++j) {
        ((uint4*)dk)[j] = k4[j];
        ((uint4*)dv)[j] = v4[j];
      }
#endif
    }
    __syncthreads();

#pragma unroll
    for (int nt = 0; nt < 4; ++nt) {
      v8f c = {};
#pragma unroll
      for (int dc = 0; dc < 2; ++dc) {
        const v16bf kf = load_b_frag_lds(&Ks[nt * 16][dc * 32], LDS_K, l16, lh);
        c = wmma_bf16(qf[dc], kf, c);
      }
#pragma unroll
      for (int v = 0; v < 8; ++v) {
        const float p = __expf(c[v] - mrow[v]) * invl[v];
        const int trow = t0 + v + lh * 8;
        atomicAdd(&avgBase[(size_t)trow * T_ + s0 + nt * 16 + l16], p * invH);
        Ps[wave][v + lh * 8][nt * 16 + l16] = (bf16)p;  // re-layout via LDS
      }
    }
    // P(16x64) @ V(64x64): K chunks of 32 over s, 4 output d-tiles of 16
#pragma unroll
    for (int kc = 0; kc < 2; ++kc) {
      const v16bf ap = load_a_frag_lds(&Ps[wave][0][kc * 32], LDS_K, l16, lh);
#pragma unroll
      for (int dt = 0; dt < 4; ++dt) {
        v16bf vf;
#pragma unroll
        for (int v = 0; v < 8; ++v) {  // B[k=s][n=d]: two s-rows per VGPR
          const int srow = kc * 32 + ((v < 4) ? 2 * v : 16 + 2 * (v - 4)) + lh * 8;
          vf[2 * v]     = Vs[srow][dt * 16 + l16];
          vf[2 * v + 1] = Vs[srow + 1][dt * 16 + l16];
        }
        o[dt] = wmma_bf16(ap, vf, o[dt]);
      }
    }
  }

  // store context (bf16) for the output projection
#pragma unroll
  for (int dt = 0; dt < 4; ++dt)
#pragma unroll
    for (int v = 0; v < 8; ++v) {
      const int trow = t0 + v + lh * 8;
      O[(size_t)trow * rowStride + headOff + dt * 16 + l16] = (bf16)o[dt][v];
    }
}

__global__ void zero_f32(float* __restrict__ p, size_t n) {
  size_t i = (size_t)blockIdx.x * blockDim.x + threadIdx.x;
  const size_t stride = (size_t)gridDim.x * blockDim.x;
  for (; i < n; i += stride) p[i] = 0.0f;
}

// ---------------------------------------------------------------------------
extern "C" void kernel_launch(void* const* d_in, const int* in_sizes, int n_in,
                              void* d_out, int out_size, void* d_ws,
                              size_t ws_size, hipStream_t stream) {
  const float* query = (const float*)d_in[0];
  const float* key   = (const float*)d_in[1];
  const float* value = (const float*)d_in[2];
  const float* wq = (const float*)d_in[3];
  const float* bq = (const float*)d_in[4];
  const float* wk = (const float*)d_in[5];
  const float* bk = (const float*)d_in[6];
  const float* wv = (const float*)d_in[7];
  const float* bv = (const float*)d_in[8];
  const float* wo = (const float*)d_in[9];
  const float* bo = (const float*)d_in[10];

  float* out = (float*)d_out;                       // (T,B,E)
  float* avg = out + (size_t)T_ * B_ * E_;          // (B,T,S)

  // workspace: bf16 q,k,v,context buffers (8 MB each, 32 MB total)
  bf16* qb = (bf16*)d_ws;
  bf16* kb = qb + (size_t)M_ * E_;
  bf16* vb = kb + (size_t)M_ * E_;
  bf16* cb = vb + (size_t)M_ * E_;

  const dim3 ggrid(E_ / BN, M_ / BM);               // (8, 32)
  const float scaling = 0.125f;                     // HD^-0.5 = 64^-0.5

  gemm_xwT_bias<float, bf16><<<ggrid, 256, 0, stream>>>(query, wq, bq, qb, scaling);
  gemm_xwT_bias<float, bf16><<<ggrid, 256, 0, stream>>>(key,   wk, bk, kb, 1.0f);
  gemm_xwT_bias<float, bf16><<<ggrid, 256, 0, stream>>>(value, wv, bv, vb, 1.0f);
  zero_f32<<<2048, 256, 0, stream>>>(avg, (size_t)B_ * T_ * T_);
  attn_fwd<<<dim3(T_ / 64, H_, B_), 128, 0, stream>>>(qb, kb, vb, cb, avg);
  gemm_xwT_bias<bf16, float><<<ggrid, 256, 0, stream>>>(cb, wo, bo, out, 1.0f);
}
